// Model_decoder_25503515803705
// MI455X (gfx1250) — matrix operand order repaired and checked by reading, not compiled, not run on hardware
//
#include <hip/hip_runtime.h>

// ---------------- CDNA5 WMMA types ----------------
typedef __attribute__((ext_vector_type(16))) __bf16 v16bf;
typedef __attribute__((ext_vector_type(8)))  float  v8f;
typedef __attribute__((ext_vector_type(8)))  int    v8i;
typedef __attribute__((ext_vector_type(4)))  int    v4i;

#define AS1 __attribute__((address_space(1)))
#define AS3 __attribute__((address_space(3)))

#if __has_builtin(__builtin_amdgcn_global_load_async_to_lds_b128)
#define HAVE_ASYNC_LD 1
#else
#define HAVE_ASYNC_LD 0
#endif
#if __has_builtin(__builtin_amdgcn_global_store_async_from_lds_b128)
#define HAVE_ASYNC_ST 1
#else
#define HAVE_ASYNC_ST 0
#endif

// Problem constants
#define Mmol 2048
#define Aat  64
#define Eed  128
#define Hd   256
#define NBT  4
#define NAT  118

// ---------------- helpers ----------------
__device__ __forceinline__ unsigned short f2bf(float f) {
  unsigned u = __builtin_bit_cast(unsigned, f);
  unsigned r = (u + 0x7FFFu + ((u >> 16) & 1u)) >> 16;   // RNE
  return (unsigned short)r;
}
__device__ __forceinline__ unsigned pk2(float a, float b) {
  return (unsigned)f2bf(a) | ((unsigned)f2bf(b) << 16);
}
__device__ __forceinline__ float bf2f(unsigned short h) {
  unsigned u = ((unsigned)h) << 16;
  return __builtin_bit_cast(float, u);
}
__device__ __forceinline__ v16bf mkfrag(v4i lo, v4i hi) {
  union { v8i i8; v16bf bf; } u;
  u.i8[0]=lo[0]; u.i8[1]=lo[1]; u.i8[2]=lo[2]; u.i8[3]=lo[3];
  u.i8[4]=hi[0]; u.i8[5]=hi[1]; u.i8[6]=hi[2]; u.i8[7]=hi[3];
  return u.bf;
}

// One 16x16 output tile, K = KT*32, A from LDS (row-major bf16, rowLen elems),
// B from packed global (tile = ct*KT+kb, 32 lanes * 32B contiguous).
// bf16 A layout (matches packed B): lanes 0-15 hold K {0..15}, lanes 16-31 hold K {16..31}.
template<int KT>
__device__ __forceinline__ v8f wmma_tile(const unsigned short* aLds, int rowLen, int rt,
                                         const unsigned short* bPacked, int ct,
                                         float bias, int lane) {
  v8f acc;
#pragma unroll
  for (int i = 0; i < 8; ++i) acc[i] = bias;
  int mrow = rt * 16 + (lane & 15);
  int kbase = (lane & 16) ? 16 : 0;
#pragma unroll
  for (int kb = 0; kb < KT; ++kb) {
    const unsigned short* pa = aLds + mrow * rowLen + kb * 32 + kbase;
    v4i alo = *(const v4i*)pa;          // K +0..7   (or +16..23 for hi lanes)
    v4i ahi = *(const v4i*)(pa + 8);    // K +8..15  (or +24..31)
    const unsigned short* pb = bPacked + ((size_t)(ct * KT + kb)) * 512 + lane * 16;
    v4i blo = *(const v4i*)pb;
    v4i bhi = *(const v4i*)(pb + 8);
    acc = __builtin_amdgcn_wmma_f32_16x16x32_bf16(
        false, mkfrag(alo, ahi), false, mkfrag(blo, bhi),
        (short)0, acc, false, false);
  }
  return acc;
}

// C/D layout: vgpr r -> row r (lanes 0-15) / row 8+r (lanes 16-31), col = lane&15
__device__ __forceinline__ void store_tile_bf16(unsigned short* d, int rowLen, int rt, int ct,
                                                v8f acc, int lane, bool relu) {
  int n = ct * 16 + (lane & 15);
  int mb = rt * 16 + ((lane & 16) ? 8 : 0);
#pragma unroll
  for (int r = 0; r < 8; ++r) {
    float v = acc[r];
    if (relu && v < 0.f) v = 0.f;
    d[(mb + r) * rowLen + n] = f2bf(v);
  }
}
__device__ __forceinline__ void store_tile_f32(float* d, int rowLen, int rt, int ct,
                                               v8f acc, int lane) {
  int n = ct * 16 + (lane & 15);
  int mb = rt * 16 + ((lane & 16) ? 8 : 0);
#pragma unroll
  for (int r = 0; r < 8; ++r) d[(mb + r) * rowLen + n] = acc[r];
}

// ---------------- prep kernels ----------------
__global__ void zero_acc_kernel(float* acc) {
  if (threadIdx.x < 4) acc[threadIdx.x] = 0.f;
}

// Repack W[K x N] (row-major fp32) into B-operand tiles:
// tile = ct*ktiles + kt; within tile: lane (0..31) * 8 dwords,
// dword d of lane l = bf16(W[k][n]) | bf16(W[k+1][n])<<16,
// k = kt*32 + (l>=16 ? 16 : 0) + 2d, n = ct*16 + (l&15)  (n>=N -> 0)
__global__ __launch_bounds__(256) void pack_w_kernel(const float* __restrict__ src,
                                                     unsigned short* __restrict__ dst,
                                                     int N, int ktiles, int ntiles) {
  int f = blockIdx.x * 256 + threadIdx.x;
  if (f >= ktiles * ntiles * 256) return;
  int tile = f >> 8, r = f & 255, lane = r >> 3, d = r & 7;
  int kt = tile % ktiles, ct = tile / ktiles;
  int k = kt * 32 + ((lane >= 16) ? 16 : 0) + 2 * d;
  int n = ct * 16 + (lane & 15);
  float v0 = (n < N) ? src[(size_t)k * N + n] : 0.f;
  float v1 = (n < N) ? src[(size_t)(k + 1) * N + n] : 0.f;
  ((unsigned*)dst)[f] = pk2(v0, v1);
}

// ---------------- proj = relu(X@Wp1+b)@Wp2+b  (bf16 out) ----------------
__global__ __launch_bounds__(256) void proj_kernel(const float* __restrict__ x,
    const unsigned short* __restrict__ w1p, const float* __restrict__ b1,
    const unsigned short* __restrict__ w2p, const float* __restrict__ b2,
    unsigned short* __restrict__ proj) {
  __shared__ __align__(16) unsigned short xs[64 * Hd];
  __shared__ __align__(16) unsigned short hs[64 * Hd];
  int tid = threadIdx.x, lane = tid & 31, wave = tid >> 5;
  size_t r0 = (size_t)blockIdx.x * 64;
  const float2* src = (const float2*)(x + r0 * Hd);
  for (int i = tid; i < 64 * Hd / 2; i += 256) {
    float2 v = src[i];
    ((unsigned*)xs)[i] = pk2(v.x, v.y);
  }
  __syncthreads();
#pragma unroll
  for (int j = 0; j < 8; ++j) {                 // 64 tiles: 4 row-tiles x 16 col-tiles
    int t = wave + 8 * j, rt = t >> 4, ct = t & 15;
    __builtin_prefetch(w1p + (size_t)(((t + 8) & 127) & 15) * 8 * 512, 0, 2);
    v8f acc = wmma_tile<8>(xs, Hd, rt, w1p, ct, b1[ct * 16 + (lane & 15)], lane);
    store_tile_bf16(hs, Hd, rt, ct, acc, lane, true);
  }
  __syncthreads();
#pragma unroll
  for (int j = 0; j < 8; ++j) {
    int t = wave + 8 * j, rt = t >> 4, ct = t & 15;
    __builtin_prefetch(w2p + (size_t)(((t + 8) & 127) & 15) * 8 * 512, 0, 2);
    v8f acc = wmma_tile<8>(hs, Hd, rt, w2p, ct, b2[ct * 16 + (lane & 15)], lane);
    store_tile_bf16(xs, Hd, rt, ct, acc, lane, false);   // xs reused as output
  }
  __syncthreads();
#if HAVE_ASYNC_ST
  // LDS -> global, 16B chunks, ASYNCcnt-tracked (no VGPR round trip)
  for (int i = tid; i < 64 * Hd / 8; i += 256) {
    __builtin_amdgcn_global_store_async_from_lds_b128(
        (AS1 v4i*)(proj + r0 * Hd + (size_t)i * 8),
        (AS3 v4i*)(xs + i * 8), 0, 0);
  }
  __asm__ volatile("s_wait_asynccnt 0" ::: "memory");
#else
  unsigned* dst = (unsigned*)(proj + r0 * Hd);
  for (int i = tid; i < 64 * Hd / 2; i += 256) dst[i] = ((unsigned*)xs)[i];
#endif
}

// ------- atom head: logits = relu(X@Wa1+b)@Wa2+b ; fused CE + acc -------
__global__ __launch_bounds__(256) void atom_kernel(const float* __restrict__ x,
    const unsigned short* __restrict__ w1p, const float* __restrict__ b1,
    const unsigned short* __restrict__ w2p, const float* __restrict__ b2,
    const int* __restrict__ tgt, float* __restrict__ accg) {
  __shared__ __align__(16) unsigned short xs[64 * Hd];
  __shared__ __align__(16) unsigned short hs[64 * Hd];
  int tid = threadIdx.x, lane = tid & 31, wave = tid >> 5;
  int r0 = blockIdx.x * 64;
  const float2* src = (const float2*)(x + (size_t)r0 * Hd);
  for (int i = tid; i < 64 * Hd / 2; i += 256) {
    float2 v = src[i];
    ((unsigned*)xs)[i] = pk2(v.x, v.y);
  }
  __syncthreads();
#pragma unroll
  for (int j = 0; j < 8; ++j) {
    int t = wave + 8 * j, rt = t >> 4, ct = t & 15;
    __builtin_prefetch(w1p + (size_t)(((t + 8) & 127) & 15) * 8 * 512, 0, 2);
    v8f acc = wmma_tile<8>(xs, Hd, rt, w1p, ct, b1[ct * 16 + (lane & 15)], lane);
    store_tile_bf16(hs, Hd, rt, ct, acc, lane, true);
  }
  __syncthreads();
  float* lg = (float*)xs;                       // 64 x 128 fp32 logits (cols >=118 junk)
#pragma unroll
  for (int j = 0; j < 4; ++j) {                 // 32 tiles: 4 row-tiles x 8 col-tiles
    int t = wave + 8 * j, rt = t >> 3, ct = t & 7;
    int n = ct * 16 + (lane & 15);
    float bias = (n < NAT) ? b2[n] : 0.f;
    v8f acc = wmma_tile<8>(hs, Hd, rt, w2p, ct, bias, lane);
    store_tile_f32(lg, 128, rt, ct, acc, lane);
  }
  __syncthreads();
  float* scratch = (float*)hs;
  if (tid < 64) {
    const float* row = lg + tid * 128;
    float mx = row[0]; int am = 0;
    for (int j = 1; j < NAT; ++j) { float v = row[j]; if (v > mx) { mx = v; am = j; } }
    float se = 0.f;
    for (int j = 0; j < NAT; ++j) se += __expf(row[j] - mx);
    int tg = tgt[r0 + tid];
    float lp = row[tg] - mx - __logf(se);
    scratch[tid] = -lp;
    scratch[64 + tid] = (am == tg) ? 1.f : 0.f;
  }
  __syncthreads();
  if (tid == 0) {
    float ls = 0.f, cs = 0.f;
    for (int j = 0; j < 64; ++j) { ls += scratch[j]; cs += scratch[64 + j]; }
    atomicAdd(&accg[2], ls);
    atomicAdd(&accg[3], cs);
  }
}

// ------- bond head: gather proj rows, relu(@Wb1+b)@Wb2+b, CE + acc -------
__global__ __launch_bounds__(256) void bond_kernel(const unsigned short* __restrict__ proj,
    const int* __restrict__ eidx, const unsigned short* __restrict__ w1p,
    const float* __restrict__ b1, const float* __restrict__ w2,
    const float* __restrict__ b2, const int* __restrict__ tgt,
    float* __restrict__ accg) {
  __shared__ __align__(16) unsigned short bs[32 * 512];  // 32 edges x 512 bf16
  __shared__ __align__(16) unsigned short hb[32 * 256];
  __shared__ float lsc[192];
  int tid = threadIdx.x, lane = tid & 31, wave = tid >> 5;
  int m = blockIdx.x >> 2;
  int e0 = (blockIdx.x & 3) * 32;
#if HAVE_ASYNC_LD
  // gather concat(proj[start], proj[end]) in 16B chunks, memory -> LDS direct
  for (int i = tid; i < 32 * 64; i += 256) {     // 2048 chunks of 16B
    int r = i >> 6, rem = i & 63, side = rem >> 5, c = rem & 31;
    int idx = eidx[m * 2 * Eed + side * Eed + (e0 + r)];
    __builtin_amdgcn_global_load_async_to_lds_b128(
        (AS1 v4i*)(proj + ((size_t)(m * Aat + idx)) * Hd + c * 8),
        (AS3 v4i*)(bs + r * 512 + side * 256 + c * 8), 0, 0);
  }
  __asm__ volatile("s_wait_asynccnt 0" ::: "memory");
#else
  for (int i = tid; i < 32 * 256; i += 256) {
    int r = i >> 8, d = i & 255;
    int side = d >> 7;
    int idx = eidx[m * 2 * Eed + side * Eed + (e0 + r)];
    unsigned v = ((const unsigned*)proj)[((size_t)(m * Aat + idx)) * (Hd / 2) + (d & 127)];
    ((unsigned*)bs)[i] = v;
  }
#endif
  __syncthreads();
#pragma unroll
  for (int j = 0; j < 4; ++j) {                 // 32 tiles: 2 row-tiles x 16 col-tiles
    int t = wave + 8 * j, rt = t >> 4, ct = t & 15;
    __builtin_prefetch(w1p + (size_t)((t + 8) & 15) * 16 * 512, 0, 2);
    v8f acc = wmma_tile<16>(bs, 512, rt, w1p, ct, b1[ct * 16 + (lane & 15)], lane);
    store_tile_bf16(hb, Hd, rt, ct, acc, lane, true);
  }
  __syncthreads();
  if (tid < 128) {                              // 256 -> 4 layer as plain dot
    int e = tid >> 2, o = tid & 3;
    float s = b2[o];
    for (int k = 0; k < Hd; ++k) s += bf2f(hb[e * Hd + k]) * w2[k * NBT + o];
    lsc[e * 4 + o] = s;
  }
  __syncthreads();
  if (tid < 32) {
    float l0 = lsc[tid * 4 + 0], l1 = lsc[tid * 4 + 1],
          l2 = lsc[tid * 4 + 2], l3 = lsc[tid * 4 + 3];
    float mx = l0; int am = 0;
    if (l1 > mx) { mx = l1; am = 1; }
    if (l2 > mx) { mx = l2; am = 2; }
    if (l3 > mx) { mx = l3; am = 3; }
    float se = __expf(l0 - mx) + __expf(l1 - mx) + __expf(l2 - mx) + __expf(l3 - mx);
    int tg = tgt[m * Eed + e0 + tid];
    float lt = (tg == 0) ? l0 : (tg == 1) ? l1 : (tg == 2) ? l2 : l3;
    lsc[128 + tid] = -(lt - mx - __logf(se));
    lsc[160 + tid] = (am == tg) ? 1.f : 0.f;
  }
  __syncthreads();
  if (tid == 0) {
    float ls = 0.f, cs = 0.f;
    for (int j = 0; j < 32; ++j) { ls += lsc[128 + j]; cs += lsc[160 + j]; }
    atomicAdd(&accg[0], ls);
    atomicAdd(&accg[1], cs);
  }
}

__global__ void finalize_kernel(const float* __restrict__ acc, float* __restrict__ out) {
  const float ne = (float)(Mmol * Eed);   // 262144
  const float na = (float)(Mmol * Aat);   // 131072
  out[0] = 0.5f * (acc[0] / ne) + 0.5f * (acc[2] / na);
  out[1] = acc[1] / ne;
  out[2] = acc[3] / na;
}

// ---------------- host launcher ----------------
extern "C" void kernel_launch(void* const* d_in, const int* in_sizes, int n_in,
                              void* d_out, int out_size, void* d_ws, size_t ws_size,
                              hipStream_t stream) {
  const float* node_rep = (const float*)d_in[0];
  const int*   edge_index = (const int*)d_in[1];
  const int*   bt_tgt = (const int*)d_in[2];
  const int*   at_tgt = (const int*)d_in[3];
  const float* Wp1 = (const float*)d_in[4];  const float* bp1 = (const float*)d_in[5];
  const float* Wp2 = (const float*)d_in[6];  const float* bp2 = (const float*)d_in[7];
  const float* Wb1 = (const float*)d_in[8];  const float* bb1 = (const float*)d_in[9];
  const float* Wb2 = (const float*)d_in[10]; const float* bb2 = (const float*)d_in[11];
  const float* Wa1 = (const float*)d_in[12]; const float* ba1 = (const float*)d_in[13];
  const float* Wa2 = (const float*)d_in[14]; const float* ba2 = (const float*)d_in[15];

  // workspace layout
  char* w = (char*)d_ws;
  float* acc = (float*)w;                                      // 4 floats
  unsigned short* wp1p = (unsigned short*)(w + 256);           // 128 KB (8x16 tiles)
  unsigned short* wp2p = wp1p + 65536;                         // 128 KB
  unsigned short* wa1p = wp2p + 65536;                         // 128 KB
  unsigned short* wa2p = wa1p + 65536;                         // 64 KB  (8x8 tiles)
  unsigned short* wb1p = wa2p + 32768;                         // 256 KB (16x16 tiles)
  unsigned short* proj = wb1p + 131072;                        // 64 MiB bf16 [131072 x 256]

  zero_acc_kernel<<<1, 32, 0, stream>>>(acc);
  pack_w_kernel<<<(8 * 16 * 256) / 256, 256, 0, stream>>>(Wp1, wp1p, Hd, 8, 16);
  pack_w_kernel<<<(8 * 16 * 256) / 256, 256, 0, stream>>>(Wp2, wp2p, Hd, 8, 16);
  pack_w_kernel<<<(8 * 16 * 256) / 256, 256, 0, stream>>>(Wa1, wa1p, Hd, 8, 16);
  pack_w_kernel<<<(8 * 8 * 256) / 256, 256, 0, stream>>>(Wa2, wa2p, NAT, 8, 8);
  pack_w_kernel<<<(16 * 16 * 256) / 256, 256, 0, stream>>>(Wb1, wb1p, Hd, 16, 16);

  proj_kernel<<<(Mmol * Aat) / 64, 256, 0, stream>>>(node_rep, wp1p, bp1, wp2p, bp2, proj);
  atom_kernel<<<(Mmol * Aat) / 64, 256, 0, stream>>>(node_rep, wa1p, ba1, wa2p, ba2, at_tgt, acc);
  bond_kernel<<<Mmol * (Eed / 32), 256, 0, stream>>>(proj, edge_index, wb1p, bb1, Wb2, bb2, bt_tgt, acc);
  finalize_kernel<<<1, 1, 0, stream>>>(acc, (float*)d_out);
}
